// QModule_64415919506110
// MI455X (gfx1250) — 8-bit matrix operand order repaired and checked by reading, compiled, not run on hardware
//
#include <hip/hip_runtime.h>

typedef __attribute__((ext_vector_type(8))) int v8i;

#define BM   128
#define BN   128
#define BK   64
#define LSTR 80

__global__ __launch_bounds__(256) void quant_x_kernel(
    const float* __restrict__ x, const float* __restrict__ in_scale,
    const float* __restrict__ act_scale, signed char* __restrict__ xq,
    long long total4, int K) {
  long long i = (long long)blockIdx.x * blockDim.x + threadIdx.x;
  if (i >= total4) return;
  const float inv = 1.0f / act_scale[0];
  const float4 v = ((const float4*)x)[i];
  const int kb = (int)((i * 4) % (long long)K);
  const float s0 = in_scale[kb + 0], s1 = in_scale[kb + 1];
  const float s2 = in_scale[kb + 2], s3 = in_scale[kb + 3];
  const float q0 = fminf(fmaxf(__builtin_rintf(v.x * s0 * inv), -127.f), 127.f);
  const float q1 = fminf(fmaxf(__builtin_rintf(v.y * s1 * inv), -127.f), 127.f);
  const float q2 = fminf(fmaxf(__builtin_rintf(v.z * s2 * inv), -127.f), 127.f);
  const float q3 = fminf(fmaxf(__builtin_rintf(v.w * s3 * inv), -127.f), 127.f);
  const int p = ((int)q0 & 255) | (((int)q1 & 255) << 8) |
                (((int)q2 & 255) << 16) | ((int)q3 << 24);
  ((int*)xq)[i] = p;
}

__global__ __launch_bounds__(256) void pack_w_kernel(
    const int* __restrict__ w, signed char* __restrict__ wq, long long total4) {
  long long i = (long long)blockIdx.x * blockDim.x + threadIdx.x;
  if (i >= total4) return;
  const int4 v = ((const int4*)w)[i];
  const int p = (v.x & 255) | ((v.y & 255) << 8) | ((v.z & 255) << 16) | (v.w << 24);
  ((int*)wq)[i] = p;
}

__global__ __launch_bounds__(256) void gemm_i8_kernel(
    const signed char* __restrict__ Aq,
    const signed char* __restrict__ Bq,
    const float* __restrict__ act_scale, const float* __restrict__ w_scale,
    const float* __restrict__ bias, float* __restrict__ out,
    int M, int N, int K) {
  __shared__ __align__(16) signed char sA[2][BM * LSTR];
  __shared__ __align__(16) signed char sB[2][BN * LSTR];

  const int tid  = threadIdx.x;
  const int lane = tid & 31;
  const int wid  = tid >> 5;
  const int wm   = wid >> 2;
  const int wn   = wid & 3;
  const int lhi  = lane >> 4;
  const int llo  = lane & 15;

  const int m0 = blockIdx.y * BM;
  const int n0 = blockIdx.x * BN;
  const long long Kll = K;

  auto stage = [&](int k0, int buf) {
#pragma unroll
    for (int i = 0; i < 2; ++i) {
      const int lin = i * 256 + tid;
      const int row = lin >> 2;
      const int c16 = lin & 3;
      const signed char* ga = Aq + (long long)(m0 + row) * Kll + k0 + c16 * 16;
      const unsigned la =
          (unsigned)(unsigned long long)&sA[buf][row * LSTR + c16 * 16];
      asm volatile("global_load_async_to_lds_b128 %0, %1, off"
                   :: "v"(la), "v"(ga) : "memory");
      const signed char* gb = Bq + (long long)(n0 + row) * Kll + k0 + c16 * 16;
      const unsigned lb =
          (unsigned)(unsigned long long)&sB[buf][row * LSTR + c16 * 16];
      asm volatile("global_load_async_to_lds_b128 %0, %1, off"
                   :: "v"(lb), "v"(gb) : "memory");
    }
  };

  const v8i vzero = {0, 0, 0, 0, 0, 0, 0, 0};
  v8i acc[4][2];
#pragma unroll
  for (int mi = 0; mi < 4; ++mi)
#pragma unroll
    for (int ni = 0; ni < 2; ++ni) acc[mi][ni] = vzero;

  stage(0, 0);
  asm volatile("s_wait_asynccnt 0" ::: "memory");
  __syncthreads();

  for (int k0 = 0; k0 < K; k0 += BK) {
    const int cur = (k0 / BK) & 1;
    if (k0 + BK < K) stage(k0 + BK, cur ^ 1);

    v8i af[4];
#pragma unroll
    for (int mi = 0; mi < 4; ++mi) {
      const signed char* pa =
          &sA[cur][(wm * 64 + mi * 16 + llo) * LSTR + (lhi << 4)];
      const int2 t0 = *(const int2*)(pa);
      const int2 t1 = *(const int2*)(pa + 8);
      const int2 t2 = *(const int2*)(pa + 32);
      const int2 t3 = *(const int2*)(pa + 40);
      v8i a;
      a[0] = t0.x; a[1] = t0.y; a[2] = t1.x; a[3] = t1.y;
      a[4] = t2.x; a[5] = t2.y; a[6] = t3.x; a[7] = t3.y;
      af[mi] = a;
    }
    v8i bf[2];
#pragma unroll
    for (int ni = 0; ni < 2; ++ni) {
      const signed char* pb =
          &sB[cur][(wn * 32 + ni * 16 + llo) * LSTR + (lhi << 4)];
      const int4 u0 = *(const int4*)(pb);
      const int4 u1 = *(const int4*)(pb + 32);
      v8i b;
      b[0] = u0.x; b[1] = u0.y; b[2] = u0.z; b[3] = u0.w;
      b[4] = u1.x; b[5] = u1.y; b[6] = u1.z; b[7] = u1.w;
      bf[ni] = b;
    }
#pragma unroll
    for (int mi = 0; mi < 4; ++mi)
#pragma unroll
      for (int ni = 0; ni < 2; ++ni)
        acc[mi][ni] = __builtin_amdgcn_wmma_i32_16x16x64_iu8(
            true, af[mi], true, bf[ni], acc[mi][ni], false, false);

    asm volatile("s_wait_asynccnt 0" ::: "memory");
    __syncthreads();
  }

  const float asc = act_scale[0];
#pragma unroll
  for (int ni = 0; ni < 2; ++ni) {
    const int n = n0 + wn * 32 + ni * 16 + llo;
    const float sc = w_scale[n] * asc;
    const float bs = bias[n];
#pragma unroll
    for (int mi = 0; mi < 4; ++mi) {
      const int mb = m0 + wm * 64 + mi * 16 + lhi * 8;
#pragma unroll
      for (int r = 0; r < 8; ++r)
        out[(long long)(mb + r) * N + n] = (float)acc[mi][ni][r] * sc + bs;
    }
  }
}

extern "C" void kernel_launch(void* const* d_in, const int* in_sizes, int n_in,
                              void* d_out, int out_size, void* d_ws, size_t ws_size,
                              hipStream_t stream) {
  const float* x          = (const float*)d_in[0];
  const float* in_scale   = (const float*)d_in[1];
  const float* act_scale  = (const float*)d_in[2];
  const int*   w_int      = (const int*)d_in[3];
  const float* w_scale    = (const float*)d_in[4];
  const float* bias       = (const float*)d_in[5];
  float*       out        = (float*)d_out;

  const long long K = in_sizes[1];
  const long long N = in_sizes[4];
  const long long M = (long long)in_sizes[0] / K;

  signed char* xq = (signed char*)d_ws;
  signed char* wq = xq + M * K;

  const long long tq = M * K / 4;
  quant_x_kernel<<<dim3((unsigned)((tq + 255) / 256)), 256, 0, stream>>>(
      x, in_scale, act_scale, xq, tq, (int)K);

  const long long tw = N * K / 4;
  pack_w_kernel<<<dim3((unsigned)((tw + 255) / 256)), 256, 0, stream>>>(
      w_int, wq, tw);

  dim3 grid((unsigned)(N / BN), (unsigned)(M / BM));
  gemm_i8_kernel<<<grid, 256, 0, stream>>>(xq, wq, act_scale, w_scale, bias,
                                           out, (int)M, (int)N, (int)K);
}
